// SimpleRecursor_40149354283017
// MI455X (gfx1250) — matrix operand order repaired and checked by reading, not compiled, not run on hardware
//
#include <hip/hip_runtime.h>

// ---------------------------------------------------------------------------
// SimpleRecursor on MI455X (gfx1250): whole binary-tree reduction per batch
// element kept inside one WGP's LDS; ALL matmuls (embed + combines) run on
// v_wmma_f32_16x16x32_f16. Embed's K=16 is zero-padded to K=32.
// Chunks reduce 128 -> 16 sub-roots (exact tiles); one global tail 256 -> 1
// pays the <16-row padding tax once instead of per-chunk.
// ---------------------------------------------------------------------------

typedef _Float16 h8  __attribute__((ext_vector_type(8)));
typedef _Float16 h16 __attribute__((ext_vector_type(16)));
typedef float    f8  __attribute__((ext_vector_type(8)));

#define B_      256
#define N_      2048
#define F_      16
#define D_      128
#define K_      256      // 2*D
#define CHUNK   128
#define NCHUNK  (N_ / CHUNK)   // 16
#define THREADS 256
#define WAVES   8

// ---- LDS layout (bytes, within extern __shared__) -------------------------
// [0)      WcT      f16 [256][128]  65536   (Wc transposed: [k][d])
// [65536)  nodes256 f16 [256][128]  65536   (16 sub-roots per chunk, tree order)
// [131072) bufA     f16 [128][128]  32768
// [163840) bufB     f16 [ 64][128]  16384
// [180224) WeT16    f16 [ 32][128]   8192   (We transposed, rows 16..31 zero)
// [188416) lv16     f16 [128][ 32]   8192   (leaf chunk, cols 16..31 zero)
// [196608) be       f32 [128]         512
// [197120) bc       f32 [128]         512
// [197632) red      f32 [128]         512
#define SMEM_BYTES 198144

// Generic WMMA GEMM level:
//   dst[m][d] = relu( sum_{k < 32*KSTEPS} src[m*KSTRIDE + k] * Wm[k][d] + bias[d] )
// A rows are contiguous runs of KSTRIDE f16 in LDS; Wm is [k][128] f16 in LDS.
// Computes/stores full 16-row M-tiles; garbage rows beyond m_out are never
// consumed by valid rows of later levels.
template<int KSTEPS, int KSTRIDE>
__device__ __forceinline__ void gemm_level(
    const _Float16* __restrict__ src, _Float16* __restrict__ dst,
    const _Float16* __restrict__ Wm, const float* __restrict__ bias_l,
    int m_out, int wave, int lane)
{
    const int mtiles  = (m_out + 15) >> 4;
    const int tiles   = mtiles * 8;           // 8 n-tiles of 16 over D=128
    const int lm      = lane & 15;
    const int halfsel = (lane < 16) ? 0 : 8;  // ISA 16-bit A-matrix K split

    for (int t = wave; t < tiles; t += WAVES) {
        const int mtile = t >> 3;
        const int ntile = t & 7;              // invariant per wave -> B frags hoisted

        f8 acc = {};
        const _Float16* arow = src + (size_t)(mtile * 16 + lm) * KSTRIDE;

        #pragma unroll
        for (int kb = 0; kb < KSTEPS; ++kb) {      // K in steps of 32
            const int kbase = kb * 32;
            // A fragment (16x32 f16): lane lm holds row lm; K layout per ISA:
            // lanes 0-15 -> K {kbase+0..7, kbase+16..23}, lanes 16-31 -> +8.
            const _Float16* ap = arow + kbase + halfsel;
            h8 alo = *(const h8*)(ap);
            h8 ahi = *(const h8*)(ap + 16);
            h16 a = __builtin_shufflevector(alo, ahi,
                      0,1,2,3,4,5,6,7,8,9,10,11,12,13,14,15);
            // B fragment (32x16 f16): lane L holds column N=L%16, K in the same
            // order as A (Wm stored [k][d] row-major in LDS).
            const _Float16* bp = Wm + (size_t)(kbase + halfsel) * D_ + ntile * 16 + lm;
            h16 b;
            #pragma unroll
            for (int s = 0; s < 8; ++s) {
                b[s]     = bp[(size_t)s * D_];
                b[s + 8] = bp[(size_t)(16 + s) * D_];
            }

            acc = __builtin_amdgcn_wmma_f32_16x16x32_f16(
                      false, a, false, b, (short)0, acc, false, false);
        }

        // C/D layout: VGPR r, lane L -> row = r + (L<16?0:8), col = L%16.
        const int col  = ntile * 16 + lm;
        const float bias = bias_l[col];
        #pragma unroll
        for (int r = 0; r < 8; ++r) {
            const int row = mtile * 16 + r + ((lane < 16) ? 0 : 8);
            float v = acc[r] + bias;
            v = v > 0.f ? v : 0.f;
            dst[(size_t)row * D_ + col] = (_Float16)v;
        }
    }
}

__global__ __launch_bounds__(THREADS)
void recursor_kernel(const float* __restrict__ leaves,
                     const float* __restrict__ We,
                     const float* __restrict__ be,
                     const float* __restrict__ Wc,
                     const float* __restrict__ bc,
                     const float* __restrict__ Wp,
                     const float* __restrict__ bp,
                     float* __restrict__ out)
{
    extern __shared__ __align__(128) char smem[];
    _Float16* WcT   = (_Float16*)(smem);
    _Float16* nodes = (_Float16*)(smem + 65536);
    _Float16* bufA  = (_Float16*)(smem + 131072);
    _Float16* bufB  = (_Float16*)(smem + 163840);
    _Float16* WeT16 = (_Float16*)(smem + 180224);
    _Float16* lv16  = (_Float16*)(smem + 188416);
    float*    be_l  = (float*)(smem + 196608);
    float*    bc_l  = (float*)(smem + 197120);
    float*    red   = (float*)(smem + 197632);

    const int tid  = threadIdx.x;
    const int wave = tid >> 5;
    const int lane = tid & 31;
    const int b    = blockIdx.x;

    // ---- Stage weights into LDS ------------------------------------------
    // Wc transposed + converted to f16 (coalesced global reads).
    for (int i = tid; i < D_ * K_; i += THREADS) {   // i = d*256 + k
        const int d = i >> 8;
        const int k = i & 255;
        WcT[(size_t)k * D_ + d] = (_Float16)Wc[i];
    }
    // Zero WeT16 (incl. K-pad rows 16..31) and lv16's pad cols 16..31.
    for (int i = tid; i < 32 * D_; i += THREADS) WeT16[i] = (_Float16)0.f;
    for (int i = tid; i < CHUNK * F_; i += THREADS) {
        const int n = i >> 4;
        const int k = 16 + (i & 15);
        lv16[n * 32 + k] = (_Float16)0.f;
    }
    if (tid < D_) { be_l[tid] = be[tid]; bc_l[tid] = bc[tid]; }
    __syncthreads();
    // WeT16[f][d] = We[d][f], rows 16..31 stay zero.
    for (int i = tid; i < D_ * F_; i += THREADS) {   // i = d*16 + f
        const int d = i >> 4;
        const int f = i & 15;
        WeT16[(size_t)f * D_ + d] = (_Float16)We[i];
    }
    __syncthreads();

    // ---- Per chunk: 128 leaves -> 16 sub-roots into nodes[ch*16 ..] -------
    for (int ch = 0; ch < NCHUNK; ++ch) {
        // Stage leaf chunk as f16 into lv16[n][0..15] (cols 16..31 remain 0).
        const float* lp = leaves + ((size_t)b * N_ + (size_t)ch * CHUNK) * F_;
        for (int i = tid; i < CHUNK * F_; i += THREADS) {
            const int n = i >> 4;
            const int f = i & 15;
            lv16[n * 32 + f] = (_Float16)lp[i];
        }
        __syncthreads();

        // Embed on WMMA (K padded 16->32): bufA = relu(lv16 @ WeT16 + be).
        gemm_level<1, 32>(lv16, bufA, WeT16, be_l, CHUNK, wave, lane);

        // Prefetch next chunk's leaves (8KB) while the tree levels compute.
        if (ch + 1 < NCHUNK) {
            const char* np = (const char*)(lp + CHUNK * F_);
            __builtin_prefetch(np + tid * 32, 0, 0);
        }
        __syncthreads();

        // 3 WMMA combine levels: 128 -> 64 -> 32 -> 16 (all tiles fully used).
        gemm_level<8, K_>(bufA, bufB, WcT, bc_l, 64, wave, lane);
        __syncthreads();
        gemm_level<8, K_>(bufB, bufA, WcT, bc_l, 32, wave, lane);
        __syncthreads();
        // m_out = 16: exactly one M-tile, stores exactly rows 0..15.
        gemm_level<8, K_>(bufA, nodes + (size_t)ch * 16 * D_, WcT, bc_l,
                          16, wave, lane);
        __syncthreads();
    }

    // ---- Global tail over 256 sub-roots: 256 -> 128 -> ... -> 1 -----------
    {
        _Float16* src = nodes;
        _Float16* dst = bufA;
        for (int m_out = 128; m_out >= 1; m_out >>= 1) {
            gemm_level<8, K_>(src, dst, WcT, bc_l, m_out, wave, lane);
            __syncthreads();
            if (m_out == 128) { src = bufA; dst = bufB; }
            else             { _Float16* t = src; src = dst; dst = t; }
        }
        // After the final swap, src points at the buffer holding the root.
        // ---- Projection: out[b] = dot(root, Wp[0]) + bp[0]  (C == 1) ------
        if (tid < D_) red[tid] = (float)src[tid] * Wp[tid];
        __syncthreads();
        if (tid == 0) {
            float s = bp[0];
            #pragma unroll 8
            for (int d = 0; d < D_; ++d) s += red[d];
            out[b] = s;
        }
    }
}

extern "C" void kernel_launch(void* const* d_in, const int* in_sizes, int n_in,
                              void* d_out, int out_size, void* d_ws, size_t ws_size,
                              hipStream_t stream) {
    (void)in_sizes; (void)n_in; (void)out_size; (void)d_ws; (void)ws_size;
    const float* leaves = (const float*)d_in[0];
    const float* We     = (const float*)d_in[1];
    const float* be     = (const float*)d_in[2];
    const float* Wc     = (const float*)d_in[3];
    const float* bc     = (const float*)d_in[4];
    const float* Wp     = (const float*)d_in[5];
    const float* bp     = (const float*)d_in[6];
    float* out = (float*)d_out;

    // Opt into >64KB dynamic LDS (CDNA5 WGP has 320KB).
    static bool attr_set = [] {
        (void)hipFuncSetAttribute((const void*)recursor_kernel,
                                  hipFuncAttributeMaxDynamicSharedMemorySize,
                                  SMEM_BYTES);
        return true;
    }();
    (void)attr_set;

    recursor_kernel<<<dim3(B_), dim3(THREADS), SMEM_BYTES, stream>>>(
        leaves, We, be, Wc, bc, Wp, bp, out);
}
